// SelfAttention_29317446762881
// MI455X (gfx1250) — compile-verified, matrix operand order checked by reading, not run on hardware
//
#include <hip/hip_runtime.h>
#include <hip/hip_bf16.h>
#include <math.h>

#define D_MODEL 1024
#define N_HEADS 16
#define D_HEAD  64
#define SEQ     2048
#define BATCH   2
#define NROWS   (BATCH * SEQ)   // 4096

typedef __bf16 bf16;
typedef __attribute__((ext_vector_type(16))) __bf16 v16bf;
typedef __attribute__((ext_vector_type(8)))  __bf16 v8bf;
typedef __attribute__((ext_vector_type(8)))  float  v8f;

// Load a 16-element A/B fragment half-pair per ISA 7.12.2:
// lane<16 -> K {kb..kb+7, kb+16..kb+23} with kb=0 ; lane>=16 -> kb=8
__device__ __forceinline__ v16bf load_frag(const bf16* base, int kb) {
    v8bf lo = *(const v8bf*)(base + kb);
    v8bf hi = *(const v8bf*)(base + kb + 16);
    return __builtin_shufflevector(lo, hi, 0,1,2,3,4,5,6,7,8,9,10,11,12,13,14,15);
}

__device__ __forceinline__ v8f wmma_bf16(v16bf a, v16bf b, v8f c) {
    return __builtin_amdgcn_wmma_f32_16x16x32_bf16(false, a, false, b, (short)0, c, false, false);
}

// Async DMA: 16B per lane, global -> LDS, tracked by ASYNCcnt
__device__ __forceinline__ void async_b128(const bf16* gsrc, bf16* ldst) {
    unsigned           l = (unsigned)(unsigned long long)ldst;
    unsigned long long g = (unsigned long long)gsrc;
    asm volatile("global_load_async_to_lds_b128 %0, %1, off" :: "v"(l), "v"(g) : "memory");
}

#define WAIT_ASYNC(n) asm volatile("s_wait_asynccnt " #n ::: "memory")
#define WAIT_DS0()    asm volatile("s_wait_dscnt 0x0" ::: "memory")

// ---------------------------------------------------------------------------
// 1) LayerNorm + temporal decay -> bf16 activations
// ---------------------------------------------------------------------------
__global__ __launch_bounds__(256)
void ln_decay_kernel(const float* __restrict__ x, const float* __restrict__ g,
                     const float* __restrict__ beta, bf16* __restrict__ xs) {
    int row = blockIdx.x;           // 0..4095
    int tid = threadIdx.x;          // 256 threads, 4 elems each
    const float4 v = ((const float4*)(x + (size_t)row * D_MODEL))[tid];
    float sum = v.x + v.y + v.z + v.w;
    float sq  = v.x*v.x + v.y*v.y + v.z*v.z + v.w*v.w;
#pragma unroll
    for (int m = 16; m >= 1; m >>= 1) {
        sum += __shfl_xor(sum, m, 32);
        sq  += __shfl_xor(sq,  m, 32);
    }
    __shared__ float rs[8], rq[8];
    int wid = tid >> 5, lid = tid & 31;
    if (lid == 0) { rs[wid] = sum; rq[wid] = sq; }
    __syncthreads();
    sum = 0.f; sq = 0.f;
#pragma unroll
    for (int i = 0; i < 8; ++i) { sum += rs[i]; sq += rq[i]; }
    float mu   = sum * (1.0f / D_MODEL);
    float var  = sq  * (1.0f / D_MODEL) - mu * mu;
    float rstd = rsqrtf(var + 1e-5f);
    int   s    = row & (SEQ - 1);
    float tw   = __powf(0.95f, (float)(SEQ - 1 - s));
    const float4 gg = ((const float4*)g)[tid];
    const float4 bb = ((const float4*)beta)[tid];
    bf16* o = xs + (size_t)row * D_MODEL + tid * 4;
    o[0] = (bf16)(((v.x - mu) * rstd * gg.x + bb.x) * tw);
    o[1] = (bf16)(((v.y - mu) * rstd * gg.y + bb.y) * tw);
    o[2] = (bf16)(((v.z - mu) * rstd * gg.z + bb.z) * tw);
    o[3] = (bf16)(((v.w - mu) * rstd * gg.w + bb.w) * tw);
}

// ---------------------------------------------------------------------------
// 2) fp32 -> bf16 weight conversion
// ---------------------------------------------------------------------------
__global__ __launch_bounds__(256)
void f32_to_bf16_kernel(const float* __restrict__ src, bf16* __restrict__ dst, int n) {
    int i = blockIdx.x * 256 + threadIdx.x;
    if (i < n) dst[i] = (bf16)src[i];
}

// ---------------------------------------------------------------------------
// GEMM mainloop: 64x128 block tile, 8 waves of 16x64 (4 WMMA per K-step),
// BK=32, double-buffered LDS fed by async global->LDS DMA, fully unrolled.
// A: [M,1024] bf16 row-major.  W: [N,1024] bf16 row-major (B = W^T).
// ---------------------------------------------------------------------------
__device__ __forceinline__ void async_tile_load(const bf16* __restrict__ A,
                                                const bf16* __restrict__ W,
                                                bf16* Abuf, bf16* Bbuf,
                                                int m0, int n0, int k0) {
    const int tid  = threadIdx.x;
    const int arow = tid >> 2, acol = (tid & 3) * 8;          // 64x32 A tile: 1 chunk/thread
    async_b128(A + (size_t)(m0 + arow) * D_MODEL + k0 + acol, Abuf + arow * 32 + acol);
#pragma unroll
    for (int j = 0; j < 2; ++j) {                              // 128x32 B tile: 2 chunks/thread
        int cid  = tid * 2 + j;
        int brow = cid >> 2, bcol = (cid & 3) * 8;
        async_b128(W + (size_t)(n0 + brow) * D_MODEL + k0 + bcol, Bbuf + brow * 32 + bcol);
    }
}

__device__ __forceinline__ void gemm_tile(const bf16* __restrict__ A,
                                          const bf16* __restrict__ W,
                                          bf16 (*Als)[64 * 32], bf16 (*Bls)[128 * 32],
                                          int m0, int n0, v8f acc[4]) {
    const int tid  = threadIdx.x;
    const int lane = tid & 31;
    const int wave = tid >> 5;
    const int mw   = (wave >> 1) * 16;     // 4 M-slots
    const int nw   = (wave & 1) * 64;      // 2 N-slots
    const int l15  = lane & 15;
    const int kb   = (lane < 16) ? 0 : 8;
    const v8f z = {0.f,0.f,0.f,0.f,0.f,0.f,0.f,0.f};
#pragma unroll
    for (int t = 0; t < 4; ++t) acc[t] = z;

    async_tile_load(A, W, Als[0], Bls[0], m0, n0, 0);          // preload buffer 0
#pragma unroll
    for (int k0 = 0; k0 < D_MODEL; k0 += 32) {
        const int cur = (k0 >> 5) & 1;
        if (k0 + 32 < D_MODEL) {
            // buffer cur^1 was fully consumed before the barrier that ended the
            // previous iteration (WMMA issue forces s_wait_dscnt on its frags)
            async_tile_load(A, W, Als[cur ^ 1], Bls[cur ^ 1], m0, n0, k0 + 32);
            WAIT_ASYNC(0x3);   // 3 newer copies outstanding -> current buffer landed
        } else {
            WAIT_ASYNC(0x0);
        }
        __syncthreads();
        // preload ALL fragments, then issue the 4-WMMA chain back-to-back
        v16bf af = load_frag(Als[cur] + (mw + l15) * 32, kb);
        v16bf bfr[4];
#pragma unroll
        for (int t = 0; t < 4; ++t)
            bfr[t] = load_frag(Bls[cur] + (nw + t * 16 + l15) * 32, kb);
#pragma unroll
        for (int t = 0; t < 4; ++t)
            acc[t] = wmma_bf16(af, bfr[t], acc[t]);
        __syncthreads();
    }
}

// ---------------------------------------------------------------------------
// 3) QKV projection: xs @ W^T + b  -> bf16 [B,H,S,dh]
// ---------------------------------------------------------------------------
__global__ __launch_bounds__(256)
void gemm_qkv_kernel(const bf16* __restrict__ xs, const bf16* __restrict__ W,
                     const float* __restrict__ bias, bf16* __restrict__ out) {
    __shared__ bf16 Als[2][64 * 32];
    __shared__ bf16 Bls[2][128 * 32];
    const int blk = blockIdx.x;            // 512 blocks
    const int m0 = (blk >> 3) * 64;
    const int n0 = (blk & 7) * 128;
    v8f acc[4];
    gemm_tile(xs, W, Als, Bls, m0, n0, acc);
    const int lane = threadIdx.x & 31;
    const int wave = threadIdx.x >> 5;
    const int mw = (wave >> 1) * 16, nw = (wave & 1) * 64;
    const int l15 = lane & 15;
    const int rofs = (lane < 16) ? 0 : 8;
#pragma unroll
    for (int t = 0; t < 4; ++t) {
        int col = n0 + nw + t * 16 + l15;
        float bcol = bias[col];
        int h = col >> 6, dd = col & 63;
#pragma unroll
        for (int r = 0; r < 8; ++r) {
            int row = m0 + mw + r + rofs;
            int b = row >> 11, s = row & (SEQ - 1);
            out[(((size_t)(b * N_HEADS + h)) * SEQ + s) * D_HEAD + dd] =
                (bf16)(acc[t][r] + bcol);
        }
    }
}

// ---------------------------------------------------------------------------
// 4) Causal flash attention: one wave per 16-query tile, 32-key tiles.
//    V tile staged by async DMA issued before the score WMMAs (overlap).
// ---------------------------------------------------------------------------
__global__ __launch_bounds__(256)
void attention_kernel(const bf16* __restrict__ q, const bf16* __restrict__ k,
                      const bf16* __restrict__ v, bf16* __restrict__ attn) {
    __shared__ bf16 Vls[8][32 * D_HEAD];   // per-wave V tile   (32 KB)
    __shared__ bf16 Pls[8][16 * 32];       // per-wave P tile   ( 8 KB)
    const int wave = threadIdx.x >> 5;
    const int lane = threadIdx.x & 31;
    const int gw = blockIdx.x * 8 + wave;  // 0..4095 q-tiles
    const int qt = gw & 127;
    const int h  = (gw >> 7) & 15;
    const int b  = gw >> 11;
    const int q0 = qt * 16;
    const bf16* qp = q + ((size_t)(b * N_HEADS + h)) * SEQ * D_HEAD;
    const bf16* kp = k + ((size_t)(b * N_HEADS + h)) * SEQ * D_HEAD;
    const bf16* vp = v + ((size_t)(b * N_HEADS + h)) * SEQ * D_HEAD;
    bf16* Vw = &Vls[wave][0];
    bf16* Pw = &Pls[wave][0];

    const int l15  = lane & 15;
    const int kb   = (lane < 16) ? 0 : 8;
    const int rofs = (lane < 16) ? 0 : 8;

    // Q A-fragments (K-dim = dh = 64 -> two K=32 fragments), straight from global
    v16bf qa[2];
#pragma unroll
    for (int f = 0; f < 2; ++f)
        qa[f] = load_frag(qp + (size_t)(q0 + l15) * D_HEAD + f * 32, kb);

    const v8f z = {0.f,0.f,0.f,0.f,0.f,0.f,0.f,0.f};
    v8f o[4] = {z, z, z, z};
    float mrow[8], lrow[8];
#pragma unroll
    for (int r = 0; r < 8; ++r) { mrow[r] = -1e30f; lrow[r] = 0.f; }

    const int kend = q0 + 15;
    for (int k0 = 0; k0 <= kend; k0 += 32) {
        // --- kick off async V-tile DMA for this key tile (overlaps QK^T) ---
        WAIT_DS0();                        // prior iteration's LDS reads complete
        {
            int vr = k0 + lane; if (vr > SEQ - 1) vr = SEQ - 1;
            const bf16* vsrc = vp + (size_t)vr * D_HEAD;
#pragma unroll
            for (int c = 0; c < 8; ++c)
                async_b128(vsrc + c * 8, Vw + lane * D_HEAD + c * 8);
        }
        // --- scores S = Q K^T : two 16x16 C tiles, K^T frags from global ---
        v8f s0 = z, s1 = z;
        int kr0 = k0 + l15;      if (kr0 > SEQ - 1) kr0 = SEQ - 1;
        int kr1 = k0 + 16 + l15; if (kr1 > SEQ - 1) kr1 = SEQ - 1;
#pragma unroll
        for (int f = 0; f < 2; ++f) {
            v16bf kf0 = load_frag(kp + (size_t)kr0 * D_HEAD + f * 32, kb);
            v16bf kf1 = load_frag(kp + (size_t)kr1 * D_HEAD + f * 32, kb);
            s0 = wmma_bf16(qa[f], kf0, s0);
            s1 = wmma_bf16(qa[f], kf1, s1);
        }
        // --- online softmax (fp32) ---
        float p0[8], p1[8], alpha[8];
#pragma unroll
        for (int r = 0; r < 8; ++r) {
            int qrow = q0 + r + rofs;
            float x0 = s0[r] * 0.125f;   // 1/sqrt(64)
            float x1 = s1[r] * 0.125f;
            if (k0 + l15      > qrow) x0 = -1e30f;
            if (k0 + 16 + l15 > qrow) x1 = -1e30f;
            float mx = fmaxf(x0, x1);
#pragma unroll
            for (int d = 8; d >= 1; d >>= 1) mx = fmaxf(mx, __shfl_xor(mx, d, 32));
            float mnew = fmaxf(mrow[r], mx);
            float a  = __expf(mrow[r] - mnew);
            float e0 = __expf(x0 - mnew);
            float e1 = __expf(x1 - mnew);
            float rsum = e0 + e1;
#pragma unroll
            for (int d = 8; d >= 1; d >>= 1) rsum += __shfl_xor(rsum, d, 32);
            lrow[r] = lrow[r] * a + rsum;
            mrow[r] = mnew;
            alpha[r] = a;
            p0[r] = e0; p1[r] = e1;
        }
#pragma unroll
        for (int t = 0; t < 4; ++t)
#pragma unroll
            for (int r = 0; r < 8; ++r) o[t][r] *= alpha[r];

        // P: C-layout -> A-fragment layout via per-wave LDS
#pragma unroll
        for (int r = 0; r < 8; ++r) {
            int prow = r + rofs;
            Pw[prow * 32 + l15]      = (bf16)p0[r];
            Pw[prow * 32 + 16 + l15] = (bf16)p1[r];
        }
        WAIT_DS0();                        // P visible to all lanes (in-order LDS)
        WAIT_ASYNC(0x0);                   // V tile landed in LDS

        // preload all fragments, then the 4-WMMA chain back-to-back
        v16bf pa = load_frag(Pw + l15 * 32, kb);
        v16bf vb[4];
#pragma unroll
        for (int t = 0; t < 4; ++t) {
            int n = t * 16 + l15;
#pragma unroll
            for (int e = 0; e < 8; ++e) {
                vb[t][e]     = Vw[(kb + e) * D_HEAD + n];
                vb[t][e + 8] = Vw[(kb + 16 + e) * D_HEAD + n];
            }
        }
#pragma unroll
        for (int t = 0; t < 4; ++t)
            o[t] = wmma_bf16(pa, vb[t], o[t]);
    }

    // finalize: O /= l, merge heads into [B,S,D] bf16
#pragma unroll
    for (int r = 0; r < 8; ++r) {
        float inv = 1.f / lrow[r];
        size_t row = (size_t)(b * SEQ + q0 + r + rofs);
#pragma unroll
        for (int t = 0; t < 4; ++t)
            attn[row * D_MODEL + h * D_HEAD + t * 16 + l15] = (bf16)(o[t][r] * inv);
    }
}

// ---------------------------------------------------------------------------
// 5) Output projection + gates + residual -> fp32
// ---------------------------------------------------------------------------
__global__ __launch_bounds__(256)
void gemm_out_kernel(const bf16* __restrict__ attn, const bf16* __restrict__ Wo,
                     const float* __restrict__ bo,   const float* __restrict__ x,
                     const float* __restrict__ volume,
                     const float* __restrict__ volatility,
                     const float* __restrict__ w_vol,
                     const float* __restrict__ w_vola,
                     float* __restrict__ out) {
    __shared__ bf16 Als[2][64 * 32];
    __shared__ bf16 Bls[2][128 * 32];
    __shared__ float gates[2][64];         // 128-wide N tile spans 2 heads
    const int blk = blockIdx.x;
    const int m0 = (blk >> 3) * 64;
    const int n0 = (blk & 7) * 128;
    v8f acc[4];
    gemm_tile(attn, Wo, Als, Bls, m0, n0, acc);

    const int tid = threadIdx.x;
    if (tid < 128) {
        int hsel = tid >> 6;               // 0/1 within this tile
        int head = (n0 >> 6) + hsel;
        int row = m0 + (tid & 63);
        int b = row >> 11, s = row & (SEQ - 1);
        float vol = volume[b * SEQ + s];
        float vla = volatility[b * SEQ + s];
        float gate = 1.f / (1.f + __expf(-vol * w_vol[0]));
        float lg[N_HEADS], mx = -1e30f;
#pragma unroll
        for (int hh = 0; hh < N_HEADS; ++hh) { lg[hh] = vla * w_vola[hh]; mx = fmaxf(mx, lg[hh]); }
        float den = 0.f;
#pragma unroll
        for (int hh = 0; hh < N_HEADS; ++hh) den += __expf(lg[hh] - mx);
        gates[hsel][tid & 63] = gate * __expf(lg[head] - mx) / den;
    }
    __syncthreads();

    const int lane = tid & 31;
    const int wave = tid >> 5;
    const int mw = (wave >> 1) * 16, nw = (wave & 1) * 64;
    const int l15 = lane & 15;
    const int rofs = (lane < 16) ? 0 : 8;
#pragma unroll
    for (int t = 0; t < 4; ++t) {
        int col = n0 + nw + t * 16 + l15;
        int hsel = (nw + t * 16) >> 6;
        float bcol = bo[col];
#pragma unroll
        for (int r = 0; r < 8; ++r) {
            int row = m0 + mw + r + rofs;
            float val = (acc[t][r] + bcol) * gates[hsel][mw + r + rofs];
            out[(size_t)row * D_MODEL + col] = val + x[(size_t)row * D_MODEL + col];
        }
    }
}

// ---------------------------------------------------------------------------
extern "C" void kernel_launch(void* const* d_in, const int* in_sizes, int n_in,
                              void* d_out, int out_size, void* d_ws, size_t ws_size,
                              hipStream_t stream) {
    const float* x          = (const float*)d_in[0];
    const float* volume     = (const float*)d_in[1];
    const float* volatility = (const float*)d_in[2];
    const float* ln_g       = (const float*)d_in[3];
    const float* ln_b       = (const float*)d_in[4];
    const float* Wq = (const float*)d_in[5];  const float* bq = (const float*)d_in[6];
    const float* Wk = (const float*)d_in[7];  const float* bk = (const float*)d_in[8];
    const float* Wv = (const float*)d_in[9];  const float* bv = (const float*)d_in[10];
    const float* Wo = (const float*)d_in[11]; const float* bo = (const float*)d_in[12];
    const float* w_vol  = (const float*)d_in[13];
    const float* w_vola = (const float*)d_in[14];

    char* ws = (char*)d_ws;
    const size_t MB = 1u << 20;
    bf16* xs   = (bf16*)(ws);                 //  8 MB  [4096,1024]
    bf16* wqb  = (bf16*)(ws +  8 * MB);       //  2 MB each
    bf16* wkb  = (bf16*)(ws + 10 * MB);
    bf16* wvb  = (bf16*)(ws + 12 * MB);
    bf16* wob  = (bf16*)(ws + 14 * MB);
    bf16* qb   = (bf16*)(ws + 16 * MB);       //  8 MB  [B,H,S,dh]
    bf16* kbuf = (bf16*)(ws + 24 * MB);
    bf16* vbuf = (bf16*)(ws + 32 * MB);
    bf16* atb  = (bf16*)(ws + 40 * MB);       //  8 MB  [B,S,D]

    ln_decay_kernel<<<NROWS, 256, 0, stream>>>(x, ln_g, ln_b, xs);

    const int wn = D_MODEL * D_MODEL;
    f32_to_bf16_kernel<<<wn / 256, 256, 0, stream>>>(Wq, wqb, wn);
    f32_to_bf16_kernel<<<wn / 256, 256, 0, stream>>>(Wk, wkb, wn);
    f32_to_bf16_kernel<<<wn / 256, 256, 0, stream>>>(Wv, wvb, wn);
    f32_to_bf16_kernel<<<wn / 256, 256, 0, stream>>>(Wo, wob, wn);

    const int gemm_blocks = (NROWS / 64) * (D_MODEL / 128);   // 512
    gemm_qkv_kernel<<<gemm_blocks, 256, 0, stream>>>(xs, wqb, bq, qb);
    gemm_qkv_kernel<<<gemm_blocks, 256, 0, stream>>>(xs, wkb, bk, kbuf);
    gemm_qkv_kernel<<<gemm_blocks, 256, 0, stream>>>(xs, wvb, bv, vbuf);

    attention_kernel<<<(BATCH * N_HEADS * (SEQ / 16)) / 8, 256, 0, stream>>>(qb, kbuf, vbuf, atb);

    gemm_out_kernel<<<gemm_blocks, 256, 0, stream>>>(atb, wob, bo, x, volume, volatility,
                                                     w_vol, w_vola, (float*)d_out);
}
